// MultiHeadCausalAttention_26199300505991
// MI455X (gfx1250) — compile-verified, matrix operand order checked by reading, not run on hardware
//
#include <hip/hip_runtime.h>

typedef __attribute__((ext_vector_type(16))) _Float16 v16h;
typedef __attribute__((ext_vector_type(8)))  _Float16 v8h;
typedef __attribute__((ext_vector_type(8)))  float    v8f;

constexpr int DM    = 2048;           // d_model
constexpr int SEQ   = 2048;           // sequence length
constexpr int BB    = 2;              // batch
constexpr int MTOT  = BB * SEQ;       // 4096 rows
constexpr int HEADS = 16;
constexpr int DK    = 128;            // head dim

// ---------------------------------------------------------------------------
// CDNA5 async global->LDS copy (GLOBAL_LOAD_ASYNC_TO_LDS_B128, ASYNCcnt).
// Probe round 2 revealed the builtin's exact signature: arg0 is a non-const
// pointer to int __attribute__((vector_size(16))) in addrspace(1) (printed as
// "__device__" in HIP mode); arg1 is the LDS side (addrspace(3)).
// ---------------------------------------------------------------------------
#if defined(__has_builtin)
#if __has_builtin(__builtin_amdgcn_global_load_async_to_lds_b128) && \
    __has_builtin(__builtin_amdgcn_s_wait_asynccnt)
#define USE_ASYNC_LDS 1
#endif
#endif

#ifdef USE_ASYNC_LDS
typedef int v4i_vs __attribute__((vector_size(4 * sizeof(int))));
typedef __attribute__((address_space(1))) v4i_vs* as1_v4i_p;
typedef __attribute__((address_space(3))) v4i_vs* as3_v4i_p;
__device__ __forceinline__ void async_cp16(void* lds, const void* g) {
  __builtin_amdgcn_global_load_async_to_lds_b128(
      (as1_v4i_p)(g), (as3_v4i_p)(lds), /*offset=*/0, /*cpol=*/0);
}
#endif

// ---------------------------------------------------------------------------
// WMMA helper: D = A(16x32 f16) * B(32x16 f16) + C(16x16 f32)
// ---------------------------------------------------------------------------
__device__ __forceinline__ v8f wmma_f16(v16h a, v16h b, v8f c) {
  return __builtin_amdgcn_wmma_f32_16x16x32_f16(
      /*neg_a=*/false, a, /*neg_b=*/false, b,
      /*c_mod=*/(short)0, c, /*reuse_a=*/false, /*reuse_b=*/false);
}

// Load one 16x32 fragment from a row-major f16 buffer.  Per the CDNA5 16-bit
// A layout: lane L holds row (L&15); half = L>>4 selects K runs
// {half*8..+7} and {16+half*8..+7} -> two 16-byte loads.
__device__ __forceinline__ v16h load_frag(const _Float16* base, int stride,
                                          int row, int k0, int lane) {
  const int half = (lane >> 4) & 1;
  const _Float16* rp = base + (size_t)row * stride + k0 + half * 8;
  v8h lo = *(const v8h*)rp;
  v8h hi = *(const v8h*)(rp + 16);
  return __builtin_shufflevector(lo, hi, 0, 1, 2, 3, 4, 5, 6, 7,
                                 8, 9, 10, 11, 12, 13, 14, 15);
}

// ---------------------------------------------------------------------------
// fp32 -> f16 conversion, 8 elements / thread
// ---------------------------------------------------------------------------
__global__ __launch_bounds__(256) void f32_to_f16(const float* __restrict__ in,
                                                  _Float16* __restrict__ out,
                                                  int n8) {
  int i = blockIdx.x * 256 + threadIdx.x;
  if (i < n8) {
    v8f v = ((const v8f*)in)[i];
    v8h o;
#pragma unroll
    for (int e = 0; e < 8; ++e) o[e] = (_Float16)v[e];
    ((v8h*)out)[i] = o;
  }
}

// ---------------------------------------------------------------------------
// C[M,N] = A[M,K] * B[N,K]^T + bias[N]      (torch Linear semantics)
// Block tile 128x128, K-step 64.  8 waves: wave (wm,wn) owns a 64x32 patch.
// Double-buffered LDS with async global->LDS staging when available.
// ---------------------------------------------------------------------------
template <bool F32OUT>
__global__ __launch_bounds__(256) void gemm_nt(const _Float16* __restrict__ A,
                                               const _Float16* __restrict__ Bm,
                                               const float* __restrict__ bias,
                                               void* __restrict__ Cv,
                                               int M, int N, int K) {
  constexpr int BKt = 64;
  constexpr int LD  = BKt + 8;  // 72 halfs -> 144B row stride, conflict-free

  const int tid  = threadIdx.x;
  const int lane = tid & 31;
  const int wid  = tid >> 5;   // 0..7
  const int wm   = wid >> 2;   // 0..1  -> +64*wm rows
  const int wn   = wid & 3;    // 0..3  -> +32*wn cols
  const int ln16 = lane & 15;
  const int half = (lane >> 4) & 1;
  const int m0   = blockIdx.y * 128;
  const int n0   = blockIdx.x * 128;

  v8f acc[8] = {};

#ifdef USE_ASYNC_LDS
  __shared__ _Float16 As[2][128 * LD];
  __shared__ _Float16 Bs[2][128 * LD];

  // issue one tile's worth of async copies (8 instructions / thread)
  auto issue = [&](int k0, int buf) {
#pragma unroll
    for (int c = 0; c < 4; ++c) {
      int idx = tid + 256 * c;   // 0..1023 ; 8 chunks of 8 halfs per row
      int row = idx >> 3;
      int ch  = idx & 7;
      async_cp16(&As[buf][row * LD + ch * 8],
                 A + (size_t)(m0 + row) * K + k0 + ch * 8);
    }
#pragma unroll
    for (int c = 0; c < 4; ++c) {
      int idx = tid + 256 * c;
      int row = idx >> 3;
      int ch  = idx & 7;
      async_cp16(&Bs[buf][row * LD + ch * 8],
                 Bm + (size_t)(n0 + row) * K + k0 + ch * 8);
    }
  };

  issue(0, 0);
  const int NT = K / BKt;
  for (int t = 0; t < NT; ++t) {
    if (t + 1 < NT) {
      issue((t + 1) * BKt, (t + 1) & 1);
      __builtin_amdgcn_s_wait_asynccnt(8);  // current tile's 8 ops done
    } else {
      __builtin_amdgcn_s_wait_asynccnt(0);
    }
    __syncthreads();
    const _Float16* Ab = As[t & 1];
    const _Float16* Bb = Bs[t & 1];
#pragma unroll
    for (int kk = 0; kk < BKt; kk += 32) {
      v16h af[4], bf[2];
#pragma unroll
      for (int mi = 0; mi < 4; ++mi)
        af[mi] = load_frag(Ab, LD, wm * 64 + mi * 16 + ln16, kk, lane);
#pragma unroll
      for (int ni = 0; ni < 2; ++ni)
        bf[ni] = load_frag(Bb, LD, wn * 32 + ni * 16 + ln16, kk, lane);
#pragma unroll
      for (int mi = 0; mi < 4; ++mi)
#pragma unroll
        for (int ni = 0; ni < 2; ++ni)
          acc[mi * 2 + ni] = wmma_f16(af[mi], bf[ni], acc[mi * 2 + ni]);
    }
    __syncthreads();
  }
#else
  __shared__ _Float16 As[128 * LD];
  __shared__ _Float16 Bs[128 * LD];
  for (int k0 = 0; k0 < K; k0 += BKt) {
#pragma unroll
    for (int c = 0; c < 4; ++c) {
      int idx = tid + 256 * c;
      int row = idx >> 3;
      int ch  = idx & 7;
      *(v8h*)&As[row * LD + ch * 8] =
          *(const v8h*)(A + (size_t)(m0 + row) * K + k0 + ch * 8);
      *(v8h*)&Bs[row * LD + ch * 8] =
          *(const v8h*)(Bm + (size_t)(n0 + row) * K + k0 + ch * 8);
    }
    __syncthreads();
#pragma unroll
    for (int kk = 0; kk < BKt; kk += 32) {
      v16h af[4], bf[2];
#pragma unroll
      for (int mi = 0; mi < 4; ++mi)
        af[mi] = load_frag(As, LD, wm * 64 + mi * 16 + ln16, kk, lane);
#pragma unroll
      for (int ni = 0; ni < 2; ++ni)
        bf[ni] = load_frag(Bs, LD, wn * 32 + ni * 16 + ln16, kk, lane);
#pragma unroll
      for (int mi = 0; mi < 4; ++mi)
#pragma unroll
        for (int ni = 0; ni < 2; ++ni)
          acc[mi * 2 + ni] = wmma_f16(af[mi], bf[ni], acc[mi * 2 + ni]);
    }
    __syncthreads();
  }
#endif

  // C/D layout: VGPR j -> row j + 8*half, col = lane&15 (within 16x16 tile)
#pragma unroll
  for (int mi = 0; mi < 4; ++mi) {
#pragma unroll
    for (int ni = 0; ni < 2; ++ni) {
      v8f a = acc[mi * 2 + ni];
      int col = n0 + wn * 32 + ni * 16 + ln16;
      float bv = bias[col];
#pragma unroll
      for (int j = 0; j < 8; ++j) {
        int row = m0 + wm * 64 + mi * 16 + j + 8 * half;
        float v = a[j] + bv;
        if (F32OUT)
          ((float*)Cv)[(size_t)row * N + col] = v;
        else
          ((_Float16*)Cv)[(size_t)row * N + col] = (_Float16)v;
      }
    }
  }
}

// ---------------------------------------------------------------------------
// Flash attention with causal mask.  Block = (b,h, 128 query rows), 8 waves,
// each wave owns 16 query rows.  Q fragments live in registers; K tile
// (64x128, async->LDS) and V tile (transposed, 128x64) staged per key step.
// ---------------------------------------------------------------------------
__global__ __launch_bounds__(256) void attn_causal(const _Float16* __restrict__ Q,
                                                   const _Float16* __restrict__ Kmat,
                                                   const _Float16* __restrict__ V,
                                                   _Float16* __restrict__ O) {
  constexpr int KSL = DK + 8;  // 136 halfs
  constexpr int VSL = 64 + 8;  // 72 halfs
  constexpr int PSL = 64 + 8;  // 72 halfs
  __shared__ _Float16 Ks[64 * KSL];       // K tile, row-major [t][d]
  __shared__ _Float16 Vs[DK * VSL];       // V tile transposed [d][t]
  __shared__ _Float16 Ps[8 * 16 * PSL];   // per-wave P relayout scratch

  const int tid  = threadIdx.x;
  const int lane = tid & 31;
  const int wid  = tid >> 5;
  const int half = (lane >> 4) & 1;
  const int ln16 = lane & 15;

  const int qt    = blockIdx.x;        // query tile (16 of them)
  const int bh    = blockIdx.y;        // b*HEADS + h  (32 of them)
  const int b     = bh / HEADS;
  const int h     = bh % HEADS;
  const int hbase = h * DK;
  const int qbase = qt * 128;
  const int wq0   = qbase + wid * 16;  // first query row of this wave

  // Q fragments for this wave's 16 rows x 128 dims (4 k-steps)
  const _Float16* Qw = Q + (size_t)(b * SEQ + wq0) * DM + hbase;
  v16h qf[4];
#pragma unroll
  for (int ks = 0; ks < 4; ++ks) qf[ks] = load_frag(Qw, DM, ln16, ks * 32, lane);

  v8f oacc[8] = {};
  float mrow[8], lrow[8];
#pragma unroll
  for (int j = 0; j < 8; ++j) { mrow[j] = -3.0e38f; lrow[j] = 0.0f; }

  const float SCL = 0.08838834764831845f;  // 1/sqrt(128)

  const int nkt = qt * 2 + 2;  // causal: key tiles 0..(qbase+127)/64
  for (int kt = 0; kt < nkt; ++kt) {
    const int kbase = kt * 64;
    const _Float16* Kg = Kmat + (size_t)(b * SEQ + kbase) * DM + hbase;
    const _Float16* Vg = V    + (size_t)(b * SEQ + kbase) * DM + hbase;

    // prefetch next tile toward the WGP (global_prefetch_b8)
    if (kt + 1 < nkt) {
      const _Float16* Kn = Kg + (size_t)64 * DM;
      const _Float16* Vn = Vg + (size_t)64 * DM;
      __builtin_prefetch(Kn + (size_t)(tid >> 2) * DM + (tid & 3) * 32, 0, 0);
      __builtin_prefetch(Vn + (size_t)(tid >> 2) * DM + (tid & 3) * 32, 0, 0);
    }

    // stage K tile (64x128) row-major
#ifdef USE_ASYNC_LDS
#pragma unroll
    for (int c = 0; c < 4; ++c) {
      int idx = tid + 256 * c;  // 0..1023 ; 16 chunks per row
      int row = idx >> 4;
      int ch  = idx & 15;
      async_cp16(&Ks[row * KSL + ch * 8], Kg + (size_t)row * DM + ch * 8);
    }
#else
#pragma unroll
    for (int c = 0; c < 4; ++c) {
      int idx = tid + 256 * c;
      int row = idx >> 4;
      int ch  = idx & 15;
      *(v8h*)&Ks[row * KSL + ch * 8] =
          *(const v8h*)(Kg + (size_t)row * DM + ch * 8);
    }
#endif
    // stage V tile transposed -> [d][t] (needs VGPR round-trip)
#pragma unroll
    for (int c = 0; c < 4; ++c) {
      int idx = tid + 256 * c;
      int row = idx >> 4;   // t
      int ch  = idx & 15;   // d chunk
      v8h v = *(const v8h*)(Vg + (size_t)row * DM + ch * 8);
#pragma unroll
      for (int e = 0; e < 8; ++e) Vs[(ch * 8 + e) * VSL + row] = v[e];
    }
#ifdef USE_ASYNC_LDS
    __builtin_amdgcn_s_wait_asynccnt(0);
#endif
    __syncthreads();

    if (kbase <= wq0 + 15) {  // wave-uniform: skip fully-masked tiles
      // S = Q K^T : 16x64 in 4 accumulators, sum over 4 k-steps of 32
      v8f sacc[4] = {};
#pragma unroll
      for (int ks = 0; ks < 4; ++ks) {
#pragma unroll
        for (int nt = 0; nt < 4; ++nt) {
          v16h bf = load_frag(Ks, KSL, nt * 16 + ln16, ks * 32, lane);
          sacc[nt] = wmma_f16(qf[ks], bf, sacc[nt]);
        }
      }
      // scale + causal mask
      const bool needMask = (kbase + 63 > wq0);
#pragma unroll
      for (int nt = 0; nt < 4; ++nt) {
        int tcol = kbase + nt * 16 + ln16;
#pragma unroll
        for (int j = 0; j < 8; ++j) {
          float s = sacc[nt][j] * SCL;
          if (needMask) {
            int qrow = wq0 + j + 8 * half;
            if (tcol > qrow) s = -3.0e38f;
          }
          sacc[nt][j] = s;
        }
      }
      // online softmax over this 16x64 block (rows j + 8*half)
      _Float16* Pw = &Ps[wid * 16 * PSL];
#pragma unroll
      for (int j = 0; j < 8; ++j) {
        float rm = fmaxf(fmaxf(sacc[0][j], sacc[1][j]),
                         fmaxf(sacc[2][j], sacc[3][j]));
#pragma unroll
        for (int sh = 1; sh < 16; sh <<= 1)
          rm = fmaxf(rm, __shfl_xor(rm, sh, 32));
        float mnew = fmaxf(mrow[j], rm);
        float scl  = __expf(mrow[j] - mnew);
        mrow[j] = mnew;
        float rs = 0.0f;
#pragma unroll
        for (int nt = 0; nt < 4; ++nt) {
          float pv = __expf(sacc[nt][j] - mnew);
          rs += pv;
          Pw[(j + 8 * half) * PSL + nt * 16 + ln16] = (_Float16)pv;
        }
#pragma unroll
        for (int sh = 1; sh < 16; sh <<= 1)
          rs += __shfl_xor(rs, sh, 32);
        lrow[j] = lrow[j] * scl + rs;
#pragma unroll
        for (int dt = 0; dt < 8; ++dt) oacc[dt][j] *= scl;
      }
      // O += P(16x64) * V(64x128)  (2 k-steps of 32 keys)
#pragma unroll
      for (int kt2 = 0; kt2 < 2; ++kt2) {
        v16h pf = load_frag(Pw, PSL, ln16, kt2 * 32, lane);
#pragma unroll
        for (int dt = 0; dt < 8; ++dt) {
          v16h vf = load_frag(Vs, VSL, dt * 16 + ln16, kt2 * 32, lane);
          oacc[dt] = wmma_f16(pf, vf, oacc[dt]);
        }
      }
    }
    __syncthreads();
  }

  // normalize and write f16 attention output at [b*SEQ+q, h*DK+d]
#pragma unroll
  for (int dt = 0; dt < 8; ++dt) {
    int col = hbase + dt * 16 + ln16;
#pragma unroll
    for (int j = 0; j < 8; ++j) {
      int row = b * SEQ + wq0 + j + 8 * half;
      float v = oacc[dt][j] / lrow[j];
      O[(size_t)row * DM + col] = (_Float16)v;
    }
  }
}

// ---------------------------------------------------------------------------
// Host-side orchestration
// ---------------------------------------------------------------------------
extern "C" void kernel_launch(void* const* d_in, const int* in_sizes, int n_in,
                              void* d_out, int out_size, void* d_ws,
                              size_t ws_size, hipStream_t stream) {
  (void)in_sizes; (void)n_in; (void)out_size; (void)ws_size;
  const float* x  = (const float*)d_in[0];
  const float* wq = (const float*)d_in[1];
  const float* bq = (const float*)d_in[2];
  const float* wk = (const float*)d_in[3];
  const float* bk = (const float*)d_in[4];
  const float* wv = (const float*)d_in[5];
  const float* bv = (const float*)d_in[6];
  const float* wo = (const float*)d_in[7];
  const float* bo = (const float*)d_in[8];

  char* p = (char*)d_ws;
  const size_t SZX = (size_t)MTOT * DM * sizeof(_Float16);  // 16 MB
  const size_t SZW = (size_t)DM * DM * sizeof(_Float16);    // 8 MB
  _Float16* xh  = (_Float16*)p; p += SZX;
  _Float16* wqh = (_Float16*)p; p += SZW;
  _Float16* wkh = (_Float16*)p; p += SZW;
  _Float16* wvh = (_Float16*)p; p += SZW;
  _Float16* woh = (_Float16*)p; p += SZW;
  _Float16* Qh  = (_Float16*)p; p += SZX;
  _Float16* Kh  = (_Float16*)p; p += SZX;
  _Float16* Vh  = (_Float16*)p; p += SZX;
  _Float16* Ah  = (_Float16*)p; p += SZX;

  const int nX8 = MTOT * DM / 8;  // 1,048,576 vectors of 8
  const int nW8 = DM * DM / 8;    //   524,288
  f32_to_f16<<<nX8 / 256, 256, 0, stream>>>(x,  xh,  nX8);
  f32_to_f16<<<nW8 / 256, 256, 0, stream>>>(wq, wqh, nW8);
  f32_to_f16<<<nW8 / 256, 256, 0, stream>>>(wk, wkh, nW8);
  f32_to_f16<<<nW8 / 256, 256, 0, stream>>>(wv, wvh, nW8);
  f32_to_f16<<<nW8 / 256, 256, 0, stream>>>(wo, woh, nW8);

  dim3 ggrid(DM / 128, MTOT / 128);  // (16, 32)
  gemm_nt<false><<<ggrid, 256, 0, stream>>>(xh, wqh, bq, Qh, MTOT, DM, DM);
  gemm_nt<false><<<ggrid, 256, 0, stream>>>(xh, wkh, bk, Kh, MTOT, DM, DM);
  gemm_nt<false><<<ggrid, 256, 0, stream>>>(xh, wvh, bv, Vh, MTOT, DM, DM);

  dim3 agrid(SEQ / 128, BB * HEADS);  // (16, 32)
  attn_causal<<<agrid, 256, 0, stream>>>(Qh, Kh, Vh, Ah);

  gemm_nt<true><<<ggrid, 256, 0, stream>>>(Ah, woh, bo, d_out, MTOT, DM, DM);
}
